// Linear_54666343743647
// MI455X (gfx1250) — matrix operand order repaired and checked by reading, not compiled, not run on hardware
//
#include <hip/hip_runtime.h>

// out = x @ dequant(W)^T + bias on gfx1250 (wave32, WMMA bf16).
// Phase 1 (memory-bound): x fp32 -> bf16, W fp32 * scale -> bf16, into d_ws.
// Phase 2 (compute-bound): bf16 WMMA GEMM with double-buffered LDS fed by
//   GLOBAL_LOAD_ASYNC_TO_LDS_B128 (ASYNCcnt), one barrier per K-tile.
// Fallback (small ws): fused single-kernel version.

typedef __attribute__((ext_vector_type(16))) __bf16 v16bf;
typedef __attribute__((ext_vector_type(8)))  float  v8f;

namespace {
constexpr int Mtot = 4096;               // B*S
constexpr int Ktot = 4096;               // IN
constexpr int Ntot = 16384;              // OUT
constexpr int SCALE_COLS = Ktot / 128;   // 32

// Phase-2 GEMM tiling
constexpr int MB = 256;                  // block M
constexpr int NB = 128;                  // block N
constexpr int KT = 64;                   // staged K per tile (2 WMMA k-steps)
constexpr int NTILES = Ktot / KT;        // 64
constexpr int LDS_U = 36;                // uint stride/row = (64 + 8 pad) bf16 / 2

constexpr size_t WS_NEED = ((size_t)Mtot * Ktot + (size_t)Ntot * Ktot) * 2;
}

__device__ __forceinline__ unsigned pack_bf16(float a, float b) {
  unsigned ua = __builtin_bit_cast(unsigned, a);
  unsigned ub = __builtin_bit_cast(unsigned, b);
  ua = (ua + 0x7FFFu + ((ua >> 16) & 1u)) >> 16;
  ub = (ub + 0x7FFFu + ((ub >> 16) & 1u)) >> 16;
  return ua | (ub << 16);
}

__device__ __forceinline__ void async_load_b128(unsigned lds_byte_off,
                                                unsigned g_byte_off,
                                                const void* base) {
  // GLOBAL_LOAD_ASYNC_TO_LDS_B128, SADDR form: 16B/lane memory -> LDS.
  asm volatile("global_load_async_to_lds_b128 %0, %1, %2"
               :: "v"(lds_byte_off), "v"(g_byte_off), "s"(base)
               : "memory");
}

__device__ __forceinline__ void wait_async0() {
#if __has_builtin(__builtin_amdgcn_s_wait_asynccnt)
  __builtin_amdgcn_s_wait_asynccnt(0);
#else
  asm volatile("s_wait_asynccnt 0x0" ::: "memory");
#endif
}

union FragBF { uint4 q[2]; v16bf v; };

// ---------------------------------------------------------------- Phase 1 ---
__global__ __launch_bounds__(256)
void cvt_x_bf16(const float* __restrict__ x, unsigned* __restrict__ xb) {
  const size_t i8 = ((size_t)blockIdx.x * 256 + threadIdx.x) * 8;   // 8 elems
  float4 a = *(const float4*)(x + i8);
  float4 b = *(const float4*)(x + i8 + 4);
  uint4 p = make_uint4(pack_bf16(a.x, a.y), pack_bf16(a.z, a.w),
                       pack_bf16(b.x, b.y), pack_bf16(b.z, b.w));
  *(uint4*)(xb + (i8 >> 1)) = p;
}

__global__ __launch_bounds__(256)
void cvt_w_bf16(const float* __restrict__ w, const float* __restrict__ scale,
                unsigned* __restrict__ wb) {
  const size_t i8 = ((size_t)blockIdx.x * 256 + threadIdx.x) * 8;
  const int row = (int)(i8 >> 12);       // / Ktot
  const int col = (int)(i8 & (Ktot - 1));
  const float s = scale[(row >> 7) * SCALE_COLS + (col >> 7)]; // 8 elems, 1 block
  float4 a = *(const float4*)(w + i8);
  float4 b = *(const float4*)(w + i8 + 4);
  uint4 p = make_uint4(pack_bf16(a.x * s, a.y * s), pack_bf16(a.z * s, a.w * s),
                       pack_bf16(b.x * s, b.y * s), pack_bf16(b.z * s, b.w * s));
  *(uint4*)(wb + (i8 >> 1)) = p;
}

// ---------------------------------------------------------------- Phase 2 ---
__global__ __launch_bounds__(256)
void gemm_bf16_wmma(const unsigned* __restrict__ xb,   // bf16 pairs [M][K]
                    const unsigned* __restrict__ wb,   // bf16 pairs [N][K]
                    const float* __restrict__ bias,
                    float* __restrict__ out)
{
  // Double-buffered tiles, async-filled. 2*(36+18)KB = 108 KB of 320 KB.
  __shared__ __attribute__((aligned(16))) unsigned As[2][MB * LDS_U];
  __shared__ __attribute__((aligned(16))) unsigned Bs[2][NB * LDS_U];

  const int tid  = threadIdx.x;
  const int lane = tid & 31;
  const int wave = tid >> 5;          // 8 waves
  const int wm   = wave >> 1;         // 0..3 : 64-row strip of M
  const int wn   = wave & 1;          // 0..1 : 64-col strip of N

  const int mBase = blockIdx.y * MB;
  const int nBase = blockIdx.x * NB;

  v8f acc[4][4] = {};                 // 64x64 wave tile: 4 M-frags x 4 N-frags

  // Issue one staged K-tile: 12 async b128 ops/lane (A: 8, B: 4), no VGPR data.
  auto issue = [&](int k0, int buf) {
#pragma unroll
    for (int t = 0; t < 8; ++t) {     // A: 256 rows x 8 chunks(8 bf16) = 2048
      const int c = tid + t * 256;
      const int row = c >> 3, kc = (c & 7) << 3;
      const unsigned lds = (unsigned)(uintptr_t)&As[buf][row * LDS_U + (kc >> 1)];
      const unsigned gof = (unsigned)(((mBase + row) * Ktot + k0 + kc) * 2);
      async_load_b128(lds, gof, xb);
    }
#pragma unroll
    for (int t = 0; t < 4; ++t) {     // B: 128 rows x 8 chunks = 1024
      const int c = tid + t * 256;
      const int row = c >> 3, kc = (c & 7) << 3;
      const unsigned lds = (unsigned)(uintptr_t)&Bs[buf][row * LDS_U + (kc >> 1)];
      const unsigned gof = (unsigned)(((nBase + row) * Ktot + k0 + kc) * 2);
      async_load_b128(lds, gof, wb);
    }
  };

  auto compute = [&](int buf) {
    const unsigned* A = As[buf];
    const unsigned* B = Bs[buf];
    const int half = lane >> 4;       // 0: lanes 0-15, 1: lanes 16-31
    const int l16  = lane & 15;
#pragma unroll
    for (int ks = 0; ks < 2; ++ks) {  // two K=32 WMMA steps per staged tile
      // B frags: lane = column N; halves hold contiguous K 0-15 / 16-31.
      FragBF bf[4];
#pragma unroll
      for (int ni = 0; ni < 4; ++ni) {
        const int brow = wn * 64 + ni * 16 + l16;
        const int kb   = ks * 32 + (half << 4);
        const uint4* p = (const uint4*)&B[brow * LDS_U + (kb >> 1)];
        bf[ni].q[0] = p[0];
        bf[ni].q[1] = p[1];
      }
#pragma unroll
      for (int mi = 0; mi < 4; ++mi) {
        // A frag: lane = row M; halves hold contiguous K 0-15 / 16-31 (+ks*32).
        const int arow = wm * 64 + mi * 16 + l16;
        const int ka   = ks * 32 + (half << 4);
        FragBF af;
        af.q[0] = *(const uint4*)&A[arow * LDS_U + (ka >> 1)];
        af.q[1] = *(const uint4*)&A[arow * LDS_U + ((ka + 8) >> 1)];
#pragma unroll
        for (int ni = 0; ni < 4; ++ni) {
          acc[mi][ni] = __builtin_amdgcn_wmma_f32_16x16x32_bf16(
              false, af.v, false, bf[ni].v, (short)0, acc[mi][ni], false, false);
        }
      }
    }
  };

  // Pipeline: wait(tile k) -> barrier (all data in; buf k+1 free) ->
  //           issue(tile k+1) -> compute(tile k). One barrier per K-tile.
  issue(0, 0);
  for (int kt = 0; kt < NTILES; ++kt) {
    wait_async0();                    // own chunks of tile kt are in LDS
    __syncthreads();                  // everyone's chunks in; k-1 compute done
    if (kt + 1 < NTILES) issue((kt + 1) * KT, (kt + 1) & 1);
    compute(kt & 1);
  }

  // Epilogue: VGPR r -> M = r (+8 for lanes 16-31), N = lane&15.
#pragma unroll
  for (int ni = 0; ni < 4; ++ni) {
    const int col = nBase + wn * 64 + ni * 16 + (lane & 15);
    const float bv = bias[col];
#pragma unroll
    for (int mi = 0; mi < 4; ++mi) {
      const int row0 = mBase + wm * 64 + mi * 16 + ((lane >> 4) << 3);
#pragma unroll
      for (int r = 0; r < 8; ++r) {
        out[(size_t)(row0 + r) * Ntot + col] = acc[mi][ni][r] + bv;
      }
    }
  }
}

// ------------------------------------------------- Fallback fused kernel ---
namespace fused {
constexpr int M_BLK = 128, N_BLK = 128, K_TILE = 32, LDSU = 20;
}

__global__ __launch_bounds__(256)
void gemm_fused_dq(const float* __restrict__ x, const float* __restrict__ w,
                   const float* __restrict__ scale, const float* __restrict__ bias,
                   float* __restrict__ out)
{
  using namespace fused;
  __shared__ __attribute__((aligned(16))) unsigned As[M_BLK * LDSU];
  __shared__ __attribute__((aligned(16))) unsigned Bs[N_BLK * LDSU];

  const int tid = threadIdx.x, lane = tid & 31, wave = tid >> 5;
  const int wm = wave >> 2, wn = wave & 3;
  const int mBase = blockIdx.y * M_BLK, nBase = blockIdx.x * N_BLK;
  const int scaleRow = nBase >> 7;

  v8f acc[4][2] = {};
  float4 ax[4], bx[4];

  auto gload = [&](int k0) {
    const float s = scale[scaleRow * SCALE_COLS + (k0 >> 7)];
#pragma unroll
    for (int t = 0; t < 4; ++t) {
      const int c = tid + t * 256, row = c >> 3, kc = (c & 7) << 2;
      ax[t] = *(const float4*)(x + (size_t)(mBase + row) * Ktot + k0 + kc);
      float4 bv = *(const float4*)(w + (size_t)(nBase + row) * Ktot + k0 + kc);
      bx[t] = make_float4(bv.x * s, bv.y * s, bv.z * s, bv.w * s);
    }
  };
  auto lstore = [&]() {
#pragma unroll
    for (int t = 0; t < 4; ++t) {
      const int c = tid + t * 256, row = c >> 3, kc = (c & 7) << 2;
      const int u = row * LDSU + (kc >> 1);
      *(uint2*)&As[u] = make_uint2(pack_bf16(ax[t].x, ax[t].y), pack_bf16(ax[t].z, ax[t].w));
      *(uint2*)&Bs[u] = make_uint2(pack_bf16(bx[t].x, bx[t].y), pack_bf16(bx[t].z, bx[t].w));
    }
  };
  auto compute = [&]() {
    const int half = lane >> 4, l16 = lane & 15;
    FragBF bf[2];
#pragma unroll
    for (int ni = 0; ni < 2; ++ni) {
      const uint4* p = (const uint4*)&Bs[(wn * 32 + ni * 16 + l16) * LDSU + (half << 3)];
      bf[ni].q[0] = p[0]; bf[ni].q[1] = p[1];
    }
#pragma unroll
    for (int mi = 0; mi < 4; ++mi) {
      const int arow = wm * 64 + mi * 16 + l16, ka = half << 4;
      FragBF af;
      af.q[0] = *(const uint4*)&As[arow * LDSU + (ka >> 1)];
      af.q[1] = *(const uint4*)&As[arow * LDSU + ((ka + 8) >> 1)];
#pragma unroll
      for (int ni = 0; ni < 2; ++ni)
        acc[mi][ni] = __builtin_amdgcn_wmma_f32_16x16x32_bf16(
            false, af.v, false, bf[ni].v, (short)0, acc[mi][ni], false, false);
    }
  };

  gload(0); lstore(); __syncthreads();
  for (int k0 = K_TILE; k0 < Ktot; k0 += K_TILE) {
    gload(k0); compute(); __syncthreads(); lstore(); __syncthreads();
  }
  compute();

#pragma unroll
  for (int ni = 0; ni < 2; ++ni) {
    const int col = nBase + wn * 32 + ni * 16 + (lane & 15);
    const float bv = bias[col];
#pragma unroll
    for (int mi = 0; mi < 4; ++mi) {
      const int row0 = mBase + wm * 64 + mi * 16 + ((lane >> 4) << 3);
#pragma unroll
      for (int r = 0; r < 8; ++r)
        out[(size_t)(row0 + r) * Ntot + col] = acc[mi][ni][r] + bv;
    }
  }
}

// ------------------------------------------------------------------ launch ---
extern "C" void kernel_launch(void* const* d_in, const int* in_sizes, int n_in,
                              void* d_out, int out_size, void* d_ws, size_t ws_size,
                              hipStream_t stream) {
  const float* x     = (const float*)d_in[0];  // [2,2048,4096]
  const float* w     = (const float*)d_in[1];  // [16384,4096]
  const float* scale = (const float*)d_in[2];  // [128,32]
  const float* bias  = (const float*)d_in[3];  // [16384]
  float* out = (float*)d_out;                  // [2,2048,16384]

  if (ws_size >= WS_NEED) {
    unsigned* xb = (unsigned*)d_ws;                          // bf16 pairs, 32 MB
    unsigned* wb = xb + ((size_t)Mtot * Ktot) / 2;           // bf16 pairs, 128 MB

    cvt_x_bf16<<<(Mtot * (size_t)Ktot) / (256 * 8), 256, 0, stream>>>(x, xb);
    cvt_w_bf16<<<((size_t)Ntot * Ktot) / (256 * 8), 256, 0, stream>>>(w, scale, wb);

    dim3 grid(Ntot / NB, Mtot / MB);                         // 128 x 16
    gemm_bf16_wmma<<<grid, 256, 0, stream>>>(xb, wb, bias, out);
  } else {
    dim3 grid(Ntot / fused::N_BLK, Mtot / fused::M_BLK);     // 128 x 32
    gemm_fused_dq<<<grid, 256, 0, stream>>>(x, w, scale, bias, out);
  }
}
